// PointNetPlusSeg_30966714204859
// MI455X (gfx1250) — matrix operand order repaired and checked by reading, not compiled, not run on hardware
//
#include <hip/hip_runtime.h>

typedef __attribute__((ext_vector_type(16))) _Float16 v16h;
typedef __attribute__((ext_vector_type(8)))  _Float16 v8h;
typedef __attribute__((ext_vector_type(8)))  float    v8f;

#define DEV __global__ __launch_bounds__(256)
#define GSTRIDE(i, n) for (int i = (int)(blockIdx.x * blockDim.x + threadIdx.x); \
                           i < (n); i += (int)(gridDim.x * blockDim.x))

// ---------------- split input (B,9,N) -> xyz (B,N,3), feats (B,N,6) ----------------
DEV void k_split(const float* __restrict__ x, float* __restrict__ xyz,
                 float* __restrict__ fe, int Btot, int N) {
  int tot = Btot * N;
  GSTRIDE(g, tot) {
    int b = g / N, n = g - b * N;
    const float* xb = x + (size_t)b * 9 * N;
    xyz[(size_t)g * 3 + 0] = xb[0 * N + n];
    xyz[(size_t)g * 3 + 1] = xb[1 * N + n];
    xyz[(size_t)g * 3 + 2] = xb[2 * N + n];
    for (int c = 0; c < 6; ++c) fe[(size_t)g * 6 + c] = xb[(3 + c) * N + n];
  }
}

// ---------------- weight convert: (Cout,Cin) f32 -> (Cout,Kpad) f16 zero-padded ----
DEV void k_convw(const float* __restrict__ W, _Float16* __restrict__ Wh,
                 int Cin, int Kpad, int n) {
  GSTRIDE(e, n) {
    int o = e / Kpad, k = e - o * Kpad;
    float v = (k < Cin) ? W[(size_t)o * Cin + k] : 0.0f;
    Wh[e] = (_Float16)v;
  }
}

// ---------------- farthest point sampling: one block per batch --------------------
DEV void k_fps(const float* __restrict__ xyz, int* __restrict__ out,
               int Nprev, int npoint) {
  const int b = blockIdx.x;
  const int t = (int)threadIdx.x;
  const int ppt = (Nprev + 255) >> 8;
  float dist[16];
  for (int i = 0; i < ppt; ++i) dist[i] = 1e10f;
  __shared__ float sv[256];
  __shared__ int   si[256];
  __shared__ float cen[3];
  const float* bx = xyz + (size_t)b * Nprev * 3;
  int far = 0;
  for (int j = 0; j < npoint; ++j) {
    if (t == 0) {
      out[b * npoint + j] = far;
      cen[0] = bx[far * 3 + 0]; cen[1] = bx[far * 3 + 1]; cen[2] = bx[far * 3 + 2];
    }
    __syncthreads();
    float cx = cen[0], cy = cen[1], cz = cen[2];
    float best = -1.0f; int bi = 0;
    for (int i = 0; i < ppt; ++i) {
      int n = i * 256 + t;
      if (n < Nprev) {
        float dx = bx[n * 3 + 0] - cx, dy = bx[n * 3 + 1] - cy, dz = bx[n * 3 + 2] - cz;
        float d = dx * dx + dy * dy + dz * dz;
        float dm = fminf(dist[i], d);
        dist[i] = dm;
        if (dm > best) { best = dm; bi = n; }
      }
    }
    sv[t] = best; si[t] = bi;
    __syncthreads();
    for (int s = 128; s > 0; s >>= 1) {
      if (t < s) {
        if (sv[t + s] > sv[t] || (sv[t + s] == sv[t] && si[t + s] < si[t])) {
          sv[t] = sv[t + s]; si[t] = si[t + s];
        }
      }
      __syncthreads();
    }
    far = si[0];
    __syncthreads();
  }
}

// ---------------- gather sampled centroids ----------------------------------------
DEV void k_gather(const float* __restrict__ xyz, const int* __restrict__ fpsIdx,
                  float* __restrict__ nxyz, int Btot, int S, int Nprev) {
  int tot = Btot * S;
  GSTRIDE(g, tot) {
    int b = g / S;
    int n = fpsIdx[g];
    const float* p = xyz + ((size_t)b * Nprev + n) * 3;
    nxyz[(size_t)g * 3 + 0] = p[0];
    nxyz[(size_t)g * 3 + 1] = p[1];
    nxyz[(size_t)g * 3 + 2] = p[2];
  }
}

// ---------------- ball query: first nsample in-radius indices, ascending ----------
DEV void k_ballquery(const float* __restrict__ nxyz, const float* __restrict__ xyz,
                     int* __restrict__ idx, int Btot, int S, int Nprev,
                     float r2, int nsample) {
  int tot = Btot * S;
  GSTRIDE(g, tot) {
    int b = g / S;
    const float* q = nxyz + (size_t)g * 3;
    float qx = q[0], qy = q[1], qz = q[2];
    const float* bx = xyz + (size_t)b * Nprev * 3;
    int* o = idx + (size_t)g * nsample;
    int cnt = 0, first = 0; bool got = false;
    for (int n = 0; n < Nprev; ++n) {
      float dx = bx[n * 3 + 0] - qx, dy = bx[n * 3 + 1] - qy, dz = bx[n * 3 + 2] - qz;
      float d = dx * dx + dy * dy + dz * dz;
      if (d <= r2) {
        if (!got) { got = true; first = n; }
        o[cnt++] = n;
        if (cnt == nsample) break;
      }
    }
    for (; cnt < nsample; ++cnt) o[cnt] = first;
  }
}

// ---------------- SA group assemble: rows=(S*nsample), K padded, f16 --------------
DEV void k_assemble_sa(const float* __restrict__ xyz, const float* __restrict__ fe,
                       const float* __restrict__ nxyz, const int* __restrict__ idx,
                       _Float16* __restrict__ X, int b, int S, int nsample,
                       int Nprev, int C, int Kpad) {
  int tot = S * nsample * Kpad;
  GSTRIDE(e, tot) {
    int row = e / Kpad, k = e - row * Kpad;
    int s = row / nsample, j = row - s * nsample;
    int n = idx[((size_t)(b * S) + s) * nsample + j];
    float v = 0.0f;
    if (k < 3)          v = xyz[((size_t)b * Nprev + n) * 3 + k] - nxyz[((size_t)b * S + s) * 3 + k];
    else if (k < 3 + C) v = fe[((size_t)b * Nprev + n) * C + (k - 3)];
    X[e] = (_Float16)v;
  }
}

// ---------------- WMMA GEMM: Y = relu(X @ W^T + bias), f16 in/out, f32 acc --------
// X (rows,K) row-major f16; W (Cout,K) row-major f16; rows%16==0, K%32==0.
// Each wave computes a 16 x (16*CT) tile: the A fragment is loaded once per K-step
// and reused by CT WMMAs (4x fewer A loads at CT=4). B-fragment = one contiguous
// 32B load per lane (weight row); A-fragment = two contiguous 16B loads per lane.
template <int CT>
DEV void k_gemm(const _Float16* __restrict__ X, const _Float16* __restrict__ Wt,
                const float* __restrict__ bias, _Float16* __restrict__ Y,
                int rows, int K, int Cout) {
  const int wave = (int)threadIdx.x >> 5;
  const int lane = (int)threadIdx.x & 31;
  const int colGroups = (Cout >> 4) / CT;
  const int rowTiles = rows >> 4;
  int tile = (int)blockIdx.x * 8 + wave;
  if (tile >= rowTiles * colGroups) return;           // uniform per wave
  const int rowT = tile / colGroups;
  const int colG = tile - rowT * colGroups;
  const int m  = lane & 15;                           // A row / B column / D column
  const int hi = lane >> 4;                           // K-half select
  const _Float16* __restrict__ arow = X + (size_t)(rowT * 16 + m) * K;
  const _Float16* brow[CT];
#pragma unroll
  for (int t = 0; t < CT; ++t)
    brow[t] = Wt + (size_t)((colG * CT + t) * 16 + m) * K;
  v8f acc[CT];
#pragma unroll
  for (int t = 0; t < CT; ++t) acc[t] = (v8f){};
  for (int k = 0; k < K; k += 32) {
    union { v16h v; v8h h[2]; } a;
    a.h[0] = *(const v8h*)(arow + k + hi * 16);       // K = hi*16 + 0..7
    a.h[1] = *(const v8h*)(arow + k + hi * 16 + 8);   // K = hi*16 + 8..15
#pragma unroll
    for (int t = 0; t < CT; ++t) {
      v16h bf = *(const v16h*)(brow[t] + k + hi * 16); // K = hi*16 + 0..15, col = m
      acc[t] = __builtin_amdgcn_wmma_f32_16x16x32_f16(
          false, a.v, false, bf, (short)0, acc[t], false, false);
    }
  }
#pragma unroll
  for (int t = 0; t < CT; ++t) {
    const int col = (colG * CT + t) * 16 + m;
    const float bv = bias[col];
    _Float16* __restrict__ yb = Y + (size_t)(rowT * 16 + hi * 8) * Cout + col;
#pragma unroll
    for (int r = 0; r < 8; ++r) {                     // VGPR r holds row hi*8 + r
      float y = fmaxf(acc[t][r] + bv, 0.0f);          // fused ReLU (all MLP layers)
      yb[(size_t)r * Cout] = (_Float16)y;
    }
  }
}

// ---------------- max over nsample -> f32 level features --------------------------
DEV void k_maxpool(const _Float16* __restrict__ Y, float* __restrict__ fe,
                   int b, int S, int nsample, int C) {
  int tot = S * C;
  GSTRIDE(e, tot) {
    int s = e / C, c = e - s * C;
    float m = -3.4e38f;
    const _Float16* p = Y + (size_t)s * nsample * C + c;
    for (int j = 0; j < nsample; ++j) m = fmaxf(m, (float)p[(size_t)j * C]);
    fe[((size_t)(b * S) + s) * C + c] = m;
  }
}

// ---------------- 3-NN + inverse-distance weights ---------------------------------
DEV void k_knn3(const float* __restrict__ txyz, const float* __restrict__ sxyz,
                int* __restrict__ ki, float* __restrict__ kw, int Btot, int Nt, int Ns) {
  int tot = Btot * Nt;
  GSTRIDE(g, tot) {
    int b = g / Nt;
    const float* q = txyz + (size_t)g * 3;
    float qx = q[0], qy = q[1], qz = q[2];
    const float* bs = sxyz + (size_t)b * Ns * 3;
    float d0 = 3.4e38f, d1 = 3.4e38f, d2 = 3.4e38f;
    int i0 = 0, i1 = 0, i2 = 0;
    for (int n = 0; n < Ns; ++n) {
      float dx = bs[n * 3 + 0] - qx, dy = bs[n * 3 + 1] - qy, dz = bs[n * 3 + 2] - qz;
      float d = dx * dx + dy * dy + dz * dz;
      if (d < d0)      { d2 = d1; i2 = i1; d1 = d0; i1 = i0; d0 = d; i0 = n; }
      else if (d < d1) { d2 = d1; i2 = i1; d1 = d; i1 = n; }
      else if (d < d2) { d2 = d; i2 = n; }
    }
    float w0 = 1.0f / (d0 + 1e-8f), w1 = 1.0f / (d1 + 1e-8f), w2 = 1.0f / (d2 + 1e-8f);
    float s = w0 + w1 + w2;
    ki[(size_t)g * 3 + 0] = i0; ki[(size_t)g * 3 + 1] = i1; ki[(size_t)g * 3 + 2] = i2;
    kw[(size_t)g * 3 + 0] = w0 / s; kw[(size_t)g * 3 + 1] = w1 / s; kw[(size_t)g * 3 + 2] = w2 / s;
  }
}

// ---------------- FP assemble: [interp(src), tgt feats] padded f16 ----------------
DEV void k_assemble_fp(const float* __restrict__ sf, const float* __restrict__ tf,
                       const int* __restrict__ ki, const float* __restrict__ kw,
                       _Float16* __restrict__ X, int b, int Nt, int Ns,
                       int Csrc, int Ctgt, int Kpad) {
  int tot = Nt * Kpad;
  GSTRIDE(e, tot) {
    int m = e / Kpad, k = e - m * Kpad;
    float v = 0.0f;
    size_t gb = (size_t)(b * Nt + m) * 3;
    if (k < Csrc) {
      v = kw[gb + 0] * sf[((size_t)b * Ns + ki[gb + 0]) * Csrc + k]
        + kw[gb + 1] * sf[((size_t)b * Ns + ki[gb + 1]) * Csrc + k]
        + kw[gb + 2] * sf[((size_t)b * Ns + ki[gb + 2]) * Csrc + k];
    } else if (k < Csrc + Ctgt) {
      v = tf[((size_t)b * Nt + m) * Ctgt + (k - Csrc)];
    }
    X[e] = (_Float16)v;
  }
}

// ---------------- f16 -> f32 stage-output copy ------------------------------------
DEV void k_h2f(const _Float16* __restrict__ Y, float* __restrict__ dst,
               int b, int Nt, int C) {
  int tot = Nt * C;
  GSTRIDE(e, tot) dst[(size_t)b * Nt * C + e] = (float)Y[e];
}

// ---------------- head: out[b,o,n] = f[n,:] . W[o,:] + bh[o] ----------------------
DEV void k_head(const _Float16* __restrict__ f, const float* __restrict__ W,
                const float* __restrict__ bh, float* __restrict__ out,
                int b, int N, int C, int O) {
  int tot = O * N;
  GSTRIDE(g, tot) {
    int o = g / N, n = g - o * N;
    float s = bh[o];
    const _Float16* fr = f + (size_t)n * C;
    const float* wr = W + (size_t)o * C;
    for (int c = 0; c < C; ++c) s += (float)fr[c] * wr[c];
    out[((size_t)b * O + o) * N + n] = s;
  }
}

// ---------------- GEMM launch helper ----------------------------------------------
static inline void launch_gemm(const _Float16* X, const _Float16* W, const float* bias,
                               _Float16* Y, int rows, int K, int Cout, hipStream_t s) {
  int rowTiles = rows / 16, colTiles = Cout / 16;
  if ((colTiles & 3) == 0) {
    int tiles = rowTiles * (colTiles / 4);
    k_gemm<4><<<dim3((unsigned)((tiles + 7) / 8)), 256, 0, s>>>(X, W, bias, Y, rows, K, Cout);
  } else {
    int tiles = rowTiles * (colTiles / 2);
    k_gemm<2><<<dim3((unsigned)((tiles + 7) / 8)), 256, 0, s>>>(X, W, bias, Y, rows, K, Cout);
  }
}

// ===================================================================================
extern "C" void kernel_launch(void* const* d_in, const int* in_sizes, int n_in,
                              void* d_out, int out_size, void* d_ws, size_t ws_size,
                              hipStream_t stream) {
  (void)out_size;
  const int Bb = 16, N = 4096, NSAMP = 32, CLS = 13;
  static const int NP[5] = {4096, 1024, 256, 64, 16};
  static const int FC[5] = {6, 64, 128, 256, 512};
  static const float RAD[4] = {0.1f, 0.2f, 0.4f, 0.8f};

  struct LayDef { int cout, cin; };
  static const LayDef LAY[22] = {
    {32, 9},   {32, 32},  {64, 32},      // sa1
    {64, 67},  {64, 64},  {128, 64},     // sa2
    {128, 131},{128, 128},{256, 128},    // sa3
    {256, 259},{256, 256},{512, 256},    // sa4
    {256, 768},{256, 256},               // fp1
    {256, 384},{256, 256},               // fp2
    {256, 320},{128, 256},               // fp3
    {128, 134},{128, 128},{128, 128},    // fp4
    {13, 128}                            // head
  };
  auto pad32 = [](int c) { return (c + 31) & ~31; };

  // -------- input pointer mapping: insertion order vs tree-flatten sorted order ---
  int wIdx[22], bIdx[22], xIdx;
  if (in_sizes[0] == Bb * 9 * N) {
    xIdx = 0;
    for (int l = 0; l < 22; ++l) { wIdx[l] = 1 + 2 * l; bIdx[l] = 2 + 2 * l; }
  } else {
    static const int order[22] = {12,13,14,15,16,17,18,19,20,21,
                                   0, 1, 2, 3, 4, 5, 6, 7, 8, 9, 10, 11};
    for (int p = 0; p < 22; ++p) { wIdx[order[p]] = 2 * p; bIdx[order[p]] = 2 * p + 1; }
    xIdx = n_in - 1;
  }

  // -------- workspace carve-up (all 256B aligned) ---------------------------------
  char* base = (char*)d_ws;
  size_t off = 0;
  auto carve = [&](size_t bytes) -> char* {
    char* p = base + off;
    off = (off + bytes + 255) & ~(size_t)255;
    return p;
  };
  float* xyzL[5]; float* featsL[5];
  for (int i = 0; i < 5; ++i) xyzL[i]   = (float*)carve((size_t)Bb * NP[i] * 3 * 4);
  for (int i = 0; i < 5; ++i) featsL[i] = (float*)carve((size_t)Bb * NP[i] * FC[i] * 4);
  float* fpf3 = (float*)carve((size_t)Bb * 64 * 256 * 4);
  float* fpf2 = (float*)carve((size_t)Bb * 256 * 256 * 4);
  float* fpf1 = (float*)carve((size_t)Bb * 1024 * 128 * 4);
  int*   fpsIdx  = (int*)carve((size_t)Bb * 1024 * 4);
  int*   ballIdx = (int*)carve((size_t)Bb * 1024 * NSAMP * 4);
  int*   knnIdx  = (int*)carve((size_t)Bb * N * 3 * 4);
  float* knnW    = (float*)carve((size_t)Bb * N * 3 * 4);
  _Float16* WhP[21];
  for (int l = 0; l < 21; ++l)
    WhP[l] = (_Float16*)carve((size_t)LAY[l].cout * pad32(LAY[l].cin) * 2);
  _Float16* bufX = (_Float16*)carve((size_t)2097152 * 2);
  _Float16* bufY = (_Float16*)carve((size_t)2097152 * 2);
  if (off > ws_size) return;

  auto blocks = [](int n) { return dim3((unsigned)((n + 255) / 256)); };

  // -------- convert all MLP weights to padded f16 ---------------------------------
  for (int l = 0; l < 21; ++l) {
    int kp = pad32(LAY[l].cin);
    int n = LAY[l].cout * kp;
    k_convw<<<blocks(n), 256, 0, stream>>>((const float*)d_in[wIdx[l]], WhP[l],
                                           LAY[l].cin, kp, n);
  }

  // -------- split input -----------------------------------------------------------
  k_split<<<blocks(Bb * N), 256, 0, stream>>>((const float*)d_in[xIdx],
                                              xyzL[0], featsL[0], Bb, N);

  // -------- SA stages -------------------------------------------------------------
  for (int st = 0; st < 4; ++st) {
    int Nprev = NP[st], S = NP[st + 1], Cprev = FC[st];
    float r2 = RAD[st] * RAD[st];
    k_fps<<<dim3((unsigned)Bb), 256, 0, stream>>>(xyzL[st], fpsIdx, Nprev, S);
    k_gather<<<blocks(Bb * S), 256, 0, stream>>>(xyzL[st], fpsIdx, xyzL[st + 1], Bb, S, Nprev);
    k_ballquery<<<blocks(Bb * S), 256, 0, stream>>>(xyzL[st + 1], xyzL[st], ballIdx,
                                                    Bb, S, Nprev, r2, NSAMP);
    int l0 = st * 3;
    int K0 = pad32(3 + Cprev);
    int rows = S * NSAMP;
    int Clast = LAY[l0 + 2].cout;
    for (int b = 0; b < Bb; ++b) {
      k_assemble_sa<<<blocks(rows * K0), 256, 0, stream>>>(
          xyzL[st], featsL[st], xyzL[st + 1], ballIdx, bufX, b, S, NSAMP, Nprev, Cprev, K0);
      _Float16* cur = bufX; _Float16* nxt = bufY;
      int K = K0;
      for (int li = 0; li < 3; ++li) {
        int l = l0 + li;
        launch_gemm(cur, WhP[l], (const float*)d_in[bIdx[l]], nxt, rows, K, LAY[l].cout, stream);
        K = LAY[l].cout;
        _Float16* t = cur; cur = nxt; nxt = t;
      }
      k_maxpool<<<blocks(S * Clast), 256, 0, stream>>>(cur, featsL[st + 1], b, S, NSAMP, Clast);
    }
  }

  // -------- FP stages -------------------------------------------------------------
  struct FpDef { int sl, tl, Csrc, Ctgt, l0, nl; const float* srcF; const float* tgtF; float* outF; };
  const FpDef FP[4] = {
    {4, 3, 512, 256, 12, 2, featsL[4], featsL[3], fpf3},
    {3, 2, 256, 128, 14, 2, fpf3,      featsL[2], fpf2},
    {2, 1, 256,  64, 16, 2, fpf2,      featsL[1], fpf1},
    {1, 0, 128,   6, 18, 3, fpf1,      featsL[0], nullptr},
  };
  for (int f = 0; f < 4; ++f) {
    const FpDef& P = FP[f];
    int Nt = NP[P.tl], Ns = NP[P.sl];
    k_knn3<<<blocks(Bb * Nt), 256, 0, stream>>>(xyzL[P.tl], xyzL[P.sl], knnIdx, knnW, Bb, Nt, Ns);
    int K0 = pad32(P.Csrc + P.Ctgt);
    int Clast = LAY[P.l0 + P.nl - 1].cout;
    for (int b = 0; b < Bb; ++b) {
      k_assemble_fp<<<blocks(Nt * K0), 256, 0, stream>>>(
          P.srcF, P.tgtF, knnIdx, knnW, bufX, b, Nt, Ns, P.Csrc, P.Ctgt, K0);
      _Float16* cur = bufX; _Float16* nxt = bufY;
      int K = K0;
      for (int li = 0; li < P.nl; ++li) {
        int l = P.l0 + li;
        launch_gemm(cur, WhP[l], (const float*)d_in[bIdx[l]], nxt, Nt, K, LAY[l].cout, stream);
        K = LAY[l].cout;
        _Float16* t = cur; cur = nxt; nxt = t;
      }
      if (P.outF) {
        k_h2f<<<blocks(Nt * Clast), 256, 0, stream>>>(cur, P.outF, b, Nt, Clast);
      } else {
        k_head<<<blocks(CLS * N), 256, 0, stream>>>(
            cur, (const float*)d_in[wIdx[21]], (const float*)d_in[bIdx[21]],
            (float*)d_out, b, N, 128, CLS);
      }
    }
  }
}
